// GraphAttention_60275571032144
// MI455X (gfx1250) — matrix operand order repaired and checked by reading, not compiled, not run on hardware
//
#include <hip/hip_runtime.h>
#include <math.h>

// ---------------------------------------------------------------------------
// Graph attention, flash-attention style, CDNA5 (gfx1250) wave32 WMMA.
//   S = (Q * 1/sqrt(512)) @ K^T  + edge_count_bias ; P = softmax(S) ; O = P @ V
// f16 inputs / f32 accumulation via v_wmma_f32_16x16x32_f16.
// Round 2: software-pipeline B-fragment LDS loads so s_wait_dscnt has slack
// instead of draining to 0 before every WMMA.
// ---------------------------------------------------------------------------

typedef __attribute__((ext_vector_type(16))) _Float16 v16h;
typedef __attribute__((ext_vector_type(8)))  _Float16 v8h;
typedef __attribute__((ext_vector_type(8)))  float    v8f;

#define N_TOK  8192
#define DIM    512
#define NEDGE  262144
#define BN     32                 // key-tile width
#define NWAVES 4                  // waves per block
#define BLOCK_ROWS (NWAVES * 16)  // 64 query rows per block
#define NTILES (N_TOK / BN)       // 256 key tiles
#define DTILES (DIM / 16)         // 32 output column tiles

static __device__ __forceinline__ v16h cat8(v8h lo, v8h hi) {
  return __builtin_shufflevector(lo, hi, 0,1,2,3,4,5,6,7,8,9,10,11,12,13,14,15);
}

// --------------------------- preprocessing kernels -------------------------

__global__ void cvt_kernel(const float* __restrict__ q, const float* __restrict__ k,
                           const float* __restrict__ v, _Float16* __restrict__ qh,
                           _Float16* __restrict__ kh, _Float16* __restrict__ vh,
                           float qscale) {
  int i = blockIdx.x * blockDim.x + threadIdx.x;
  if (i < N_TOK * DIM) {
    qh[i] = (_Float16)(q[i] * qscale);   // fold 1/sqrt(d) into Q
    kh[i] = (_Float16)k[i];
    vh[i] = (_Float16)v[i];
  }
}

__global__ void zero_counts(int* __restrict__ counts) {
  int i = blockIdx.x * blockDim.x + threadIdx.x;
  if (i < N_TOK) counts[i] = 0;
}

__global__ void count_edges(const int* __restrict__ esrc, int* __restrict__ counts) {
  int i = blockIdx.x * blockDim.x + threadIdx.x;
  if (i < NEDGE) atomicAdd(&counts[esrc[i]], 1);
}

// single-block exclusive scan of 8192 per-row counts -> CSR row offsets
__global__ void scan_offsets(const int* __restrict__ counts, int* __restrict__ roff,
                             int* __restrict__ cursor) {
  __shared__ int ssum[1024];
  const int tid = threadIdx.x;
  int c[8]; int s = 0;
#pragma unroll
  for (int i = 0; i < 8; ++i) { c[i] = counts[tid * 8 + i]; s += c[i]; }
  ssum[tid] = s;
  __syncthreads();
  for (int st = 1; st < 1024; st <<= 1) {
    int add = (tid >= st) ? ssum[tid - st] : 0;
    __syncthreads();
    ssum[tid] += add;
    __syncthreads();
  }
  int run = (tid == 0) ? 0 : ssum[tid - 1];
#pragma unroll
  for (int i = 0; i < 8; ++i) {
    roff[tid * 8 + i] = run; cursor[tid * 8 + i] = run; run += c[i];
  }
  if (tid == 1023) roff[N_TOK] = run;
}

__global__ void fill_csr(const int* __restrict__ esrc, const int* __restrict__ edst,
                         int* __restrict__ cursor, int* __restrict__ edsort) {
  int i = blockIdx.x * blockDim.x + threadIdx.x;
  if (i < NEDGE) {
    int pos = atomicAdd(&cursor[esrc[i]], 1);
    edsort[pos] = edst[i];   // unsorted within row: bias is a sum, order-invariant
  }
}

// ------------------------------- main kernel -------------------------------

__global__ __launch_bounds__(NWAVES * 32, 1)
void flash_kernel(const _Float16* __restrict__ qh, const _Float16* __restrict__ kh,
                  const _Float16* __restrict__ vh, const int* __restrict__ roff,
                  const int* __restrict__ edsort, float* __restrict__ out) {
  __shared__ _Float16 sK [BN * DIM];          // 32 KB, K tile row-major
  __shared__ _Float16 sVT[DIM * BN];          // 32 KB, V tile transposed [d][key]
  __shared__ float    sScr[NWAVES * 16 * BN]; // 8 KB per-wave bias / P bounce

  const int tid  = threadIdx.x;
  const int lane = tid & 31;
  const int wave = tid >> 5;
  const int hl   = lane >> 4;    // half-lane group (ISA fragment layouts)
  const int ln   = lane & 15;
  const int rowbase = blockIdx.x * BLOCK_ROWS + wave * 16;

  // ---- preload this wave's 16x512 Q strip as 16 A-fragments (16x32 f16) ----
  // A layout: lane holds row M=ln; v16h = K[hl*8 .. +7] then K[16+hl*8 .. +7]
  v16h qa[16];
  const _Float16* qrow = qh + (size_t)(rowbase + ln) * DIM;
#pragma unroll
  for (int c = 0; c < 16; ++c) {
    const int kb = c * 32 + hl * 8;
    qa[c] = cat8(*(const v8h*)(qrow + kb), *(const v8h*)(qrow + kb + 16));
  }

  v8f o[DTILES];
  v8f zero = {};
#pragma unroll
  for (int t = 0; t < DTILES; ++t) o[t] = zero;
  float mrow[8], lrow[8];
#pragma unroll
  for (int i = 0; i < 8; ++i) { mrow[i] = -1e30f; lrow[i] = 0.0f; }

  float*    bias = sScr + wave * 16 * BN;   // 16x32 f32, wave-private
  _Float16* pbuf = (_Float16*)bias;         // reused as 16x32 f16 P bounce

  // B-fragment loaders (LDS -> fragment registers)
  auto loadKB = [&](int c, int ntile) -> v16h {
    // B layout (32x16 f16): lane holds col N=ln; K[hl*8 .. +7] then K[16+hl*8 .. +7]
    const int kb = c * 32 + hl * 8;
    const _Float16* kr = sK + (size_t)(ntile * 16 + ln) * DIM + kb;
    return cat8(*(const v8h*)kr, *(const v8h*)(kr + 16));
  };
  auto loadVB = [&](int tn) -> v16h {
    const _Float16* vc = sVT + (size_t)(tn * 16 + ln) * BN + hl * 8;
    return cat8(*(const v8h*)vc, *(const v8h*)(vc + 16));
  };

  for (int t = 0; t < NTILES; ++t) {
    const int c0 = t * BN;
    __syncthreads();   // previous iteration done reading sK/sVT

    // cooperative load K tile (row-major) and V tile (transposed)
    for (int u = tid; u < BN * DIM / 8; u += NWAVES * 32)
      *(v8h*)(sK + u * 8) = *(const v8h*)(kh + (size_t)c0 * DIM + u * 8);
    for (int u = tid; u < BN * DIM / 8; u += NWAVES * 32) {
      const int key = (u * 8) / DIM, d0 = (u * 8) % DIM;
      v8h vv = *(const v8h*)(vh + (size_t)(c0 + key) * DIM + d0);
#pragma unroll
      for (int e = 0; e < 8; ++e) sVT[(d0 + e) * BN + key] = vv[e];
    }
    if (t + 1 < NTILES) {  // hint next tile into cache
      __builtin_prefetch(kh + (size_t)(c0 + BN) * DIM + tid * 16, 0, 1);
      __builtin_prefetch(vh + (size_t)(c0 + BN) * DIM + tid * 16, 0, 1);
    }
    for (int u = lane; u < 16 * BN; u += 32) bias[u] = 0.0f;
    __syncthreads();

    // edge bias: one lane per query row scans its CSR edge list
    if (lane < 16) {
      const int r  = rowbase + lane;
      float* brow  = bias + lane * BN;
      const int e1 = roff[r + 1];
      for (int e = roff[r]; e < e1; ++e) {
        const int d = edsort[e] - c0;
        if ((unsigned)d < (unsigned)BN) brow[d] += 1.0f;  // duplicates accumulate
      }
    }
    __syncthreads();

    // ---- S tile: 16x32, contraction over D=512 in 16 WMMA k-steps --------
    // Software-pipelined: fetch k-step c+1's B fragments before WMMAs of c.
    v8f s0 = zero, s1 = zero;
    v16h b0 = loadKB(0, 0), b1 = loadKB(0, 1);
#pragma unroll
    for (int c = 0; c < 16; ++c) {
      v16h nb0, nb1;
      if (c < 15) { nb0 = loadKB(c + 1, 0); nb1 = loadKB(c + 1, 1); }
      s0 = __builtin_amdgcn_wmma_f32_16x16x32_f16(false, qa[c], false, b0,
                                                  (short)0, s0, false, false);
      s1 = __builtin_amdgcn_wmma_f32_16x16x32_f16(false, qa[c], false, b1,
                                                  (short)0, s1, false, false);
      if (c < 15) { b0 = nb0; b1 = nb1; }
    }

    // ---- bias add + online softmax (C layout: m = i + 8*hl, n = ln) ------
    float alpha[8];
#pragma unroll
    for (int i = 0; i < 8; ++i) {
      const int m = i + 8 * hl;
      float a0 = s0[i] + bias[m * BN + ln];
      float a1 = s1[i] + bias[m * BN + 16 + ln];
      float v = fmaxf(a0, a1);
#pragma unroll
      for (int msk = 1; msk <= 8; msk <<= 1) v = fmaxf(v, __shfl_xor(v, msk, 32));
      const float mnew = fmaxf(mrow[i], v);
      alpha[i] = __expf(mrow[i] - mnew);
      mrow[i]  = mnew;
      const float p0 = __expf(a0 - mnew);
      const float p1 = __expf(a1 - mnew);
      s0[i] = p0; s1[i] = p1;
      float rs = p0 + p1;
#pragma unroll
      for (int msk = 1; msk <= 8; msk <<= 1) rs += __shfl_xor(rs, msk, 32);
      lrow[i] = lrow[i] * alpha[i] + rs;
    }
#pragma unroll
    for (int tn = 0; tn < DTILES; ++tn) {
#pragma unroll
      for (int i = 0; i < 8; ++i) o[tn][i] *= alpha[i];
    }

    // ---- transpose P (C layout) -> A fragment via wave-private LDS -------
#pragma unroll
    for (int i = 0; i < 8; ++i) {
      const int m = i + 8 * hl;
      pbuf[m * BN + ln]      = (_Float16)s0[i];
      pbuf[m * BN + 16 + ln] = (_Float16)s1[i];
    }
    asm volatile("s_wait_dscnt 0" ::: "memory");  // wave-private: DS in-order, no barrier
    const _Float16* prow = pbuf + ln * BN;
    v16h pa = cat8(*(const v8h*)(prow + hl * 8), *(const v8h*)(prow + 16 + hl * 8));

    // ---- O += P(16x32) @ V(32x512): 32 WMMAs, K=32 covered in one step ---
    // Software-pipelined one tile ahead.
    v16h vb = loadVB(0);
#pragma unroll
    for (int tn = 0; tn < DTILES; ++tn) {
      v16h nvb;
      if (tn < DTILES - 1) nvb = loadVB(tn + 1);
      o[tn] = __builtin_amdgcn_wmma_f32_16x16x32_f16(false, pa, false, vb,
                                                     (short)0, o[tn], false, false);
      if (tn < DTILES - 1) vb = nvb;
    }
  }

  // ---- epilogue: divide by row sums, store f32 -----------------------------
  float inv[8];
#pragma unroll
  for (int i = 0; i < 8; ++i) inv[i] = 1.0f / lrow[i];
#pragma unroll
  for (int tn = 0; tn < DTILES; ++tn) {
#pragma unroll
    for (int i = 0; i < 8; ++i) {
      const int m = i + 8 * hl;
      out[(size_t)(rowbase + m) * DIM + tn * 16 + ln] = o[tn][i] * inv[i];
    }
  }
}

// ------------------------------- launcher ----------------------------------

extern "C" void kernel_launch(void* const* d_in, const int* in_sizes, int n_in,
                              void* d_out, int out_size, void* d_ws, size_t ws_size,
                              hipStream_t stream) {
  (void)in_sizes; (void)n_in; (void)out_size; (void)ws_size;
  const float* q    = (const float*)d_in[0];
  const float* k    = (const float*)d_in[1];
  const float* v    = (const float*)d_in[2];
  const int*   esrc = (const int*)d_in[3];
  const int*   edst = (const int*)d_in[4];
  float* out = (float*)d_out;

  char* ws = (char*)d_ws;
  size_t off = 0;
  auto take = [&](size_t bytes) {
    char* p = ws + off;
    off = (off + bytes + 255) & ~(size_t)255;
    return p;
  };
  _Float16* qh    = (_Float16*)take((size_t)N_TOK * DIM * 2);
  _Float16* kh    = (_Float16*)take((size_t)N_TOK * DIM * 2);
  _Float16* vh    = (_Float16*)take((size_t)N_TOK * DIM * 2);
  int* counts     = (int*)take((size_t)N_TOK * 4);
  int* roff       = (int*)take((size_t)(N_TOK + 1) * 4);
  int* cursor     = (int*)take((size_t)N_TOK * 4);
  int* edsort     = (int*)take((size_t)NEDGE * 4);

  const float qscale = 1.0f / sqrtf((float)DIM);
  cvt_kernel  <<<(N_TOK * DIM + 255) / 256, 256, 0, stream>>>(q, k, v, qh, kh, vh, qscale);
  zero_counts <<<(N_TOK + 255) / 256,       256, 0, stream>>>(counts);
  count_edges <<<(NEDGE + 255) / 256,       256, 0, stream>>>(esrc, counts);
  scan_offsets<<<1, 1024, 0, stream>>>(counts, roff, cursor);
  fill_csr    <<<(NEDGE + 255) / 256,       256, 0, stream>>>(esrc, edst, cursor, edsort);
  flash_kernel<<<N_TOK / BLOCK_ROWS, NWAVES * 32, 0, stream>>>(qh, kh, vh, roff, edsort, out);
}
